// vfnet_87548613362183
// MI455X (gfx1250) — compile-verified, matrix operand order checked by reading, not run on hardware
//
#include <hip/hip_runtime.h>
#include <hip/hip_bf16.h>

typedef __attribute__((ext_vector_type(16))) __bf16    v16bf;
typedef __attribute__((ext_vector_type(8)))  float     v8f;
typedef __attribute__((ext_vector_type(4)))  unsigned  v4u;
typedef __attribute__((ext_vector_type(8)))  int       v8i;
typedef __attribute__((ext_vector_type(4)))  int       v4i;

// ---------------------------------------------------------------------------
// Implicit-GEMM conv (NCHW, OIHW weights) on the WMMA pipe.
//   M = Cout tile (16), N = output-pixel tile (16/wave, 8 waves/block),
//   K = Cin*KS*KS in chunks of 32, bf16 operands, f32 accum.
// Weights are staged 16x32 per chunk into double-buffered LDS by the Tensor
// Data Mover (tensor_load_to_lds), pipelined one chunk ahead and synchronized
// with s_wait_tensorcnt. TDM zero-fills the K tail / Cout tail via the D#
// tensor-dim OOB rule, so A needs no predication at all. B (im2col) uses
// clamped always-in-bounds addresses + one cndmask per element: no exec-mask
// branches anywhere in the K loop, EXEC is all-ones at every v_wmma.
// ---------------------------------------------------------------------------
template<int KS>
__global__ __launch_bounds__(256)
void conv2d_wmma(const float* __restrict__ in, long long in_bstride,
                 const float* __restrict__ w, const float* __restrict__ bias,
                 float* __restrict__ out, long long out_bstride,
                 int Cin, int Cout, int Hin, int Win, int Hout, int Wout,
                 int stride, int leaky)
{
    constexpr int KK2 = KS * KS;
    // Only __shared__ object in this kernel -> LDS offset 0.
    __shared__ __align__(16) float Abuf[2][16 * 32];

    const int lane = threadIdx.x & 31;
    const int wv   = threadIdx.x >> 5;        // wave in block: 0..7
    const int half = lane >> 4;
    const int l16  = lane & 15;
    const int b    = blockIdx.z;
    const int m0   = blockIdx.y * 16;         // cout tile base
    const int n    = (blockIdx.x * 8 + wv) * 16 + l16;
    const int npx  = Hout * Wout;
    const bool nV  = (n < npx);
    const int nc   = nV ? n : (npx - 1);
    const int oy   = nc / Wout;
    const int ox   = nc - oy * Wout;
    const int K    = Cin * KK2;
    const int pad  = (KS - 1) >> 1;
    const int nchunks = (K + 31) >> 5;

    const float* inb = in + (long long)b * in_bstride;

    v8f acc = {0.f, 0.f, 0.f, 0.f, 0.f, 0.f, 0.f, 0.f};

    // ---- TDM: DMA weight tile rows [m0..m0+15], cols [32c..32c+31] into LDS.
    auto issue_tdm = [&](int chunk, int bufidx) {
        unsigned long long ga =
            (unsigned long long)(const void*)(w + (size_t)m0 * K + (size_t)chunk * 32);
        unsigned lds_off = (unsigned)(bufidx * (16 * 32 * 4));
        unsigned td0 = (unsigned)(K - chunk * 32);   // remaining cols: OOB cols zero-fill
        unsigned td1 = (unsigned)(Cout - m0);        // remaining rows: OOB rows zero-fill
        unsigned long long s0 = (unsigned long long)K;  // row stride (elements)

        v4u g0;
        g0[0] = 1u;                                             // count=1
        g0[1] = lds_off;                                        // lds_addr
        g0[2] = (unsigned)ga;                                   // global_addr[31:0]
        g0[3] = (unsigned)((ga >> 32) & 0x1FFFFFFull) | (2u << 30); // addr[56:32] | type=2
        v8i g1;
        g1[0] = (int)(2u << 16);                                // data_size = 4B
        g1[1] = (int)((td0 & 0xFFFFu) << 16);                   // tensor_dim0[15:0]
        g1[2] = (int)((td0 >> 16) | ((td1 & 0xFFFFu) << 16));   // dim0[31:16] | dim1[15:0]
        g1[3] = (int)((td1 >> 16) | (32u << 16));               // dim1[31:16] | tile_dim0=32
        g1[4] = (int)16u;                                       // tile_dim1=16, tile_dim2=0
        g1[5] = (int)(unsigned)(s0 & 0xFFFFFFFFull);            // dim0_stride[31:0]
        g1[6] = (int)(unsigned)((s0 >> 32) & 0xFFFFull);        // dim0_stride[47:32]
        g1[7] = 0;
        v4i g2 = {0, 0, 0, 0}, g3 = {0, 0, 0, 0};
#if __clang_major__ >= 23
        v8i g4 = {0, 0, 0, 0, 0, 0, 0, 0};
        __builtin_amdgcn_tensor_load_to_lds(g0, g1, g2, g3, g4, 0);
#else
        __builtin_amdgcn_tensor_load_to_lds(g0, g1, g2, g3, 0);
#endif
    };

    if (wv == 0) issue_tdm(0, 0);

    for (int c = 0; c < nchunks; ++c) {
        const int k0 = c << 5;
        if (wv == 0) {
            if (c + 1 < nchunks) {
                issue_tdm(c + 1, (c + 1) & 1);                 // pipeline next chunk
                __builtin_amdgcn_s_wait_tensorcnt(1);          // chunk c complete
            } else {
                __builtin_amdgcn_s_wait_tensorcnt(0);
            }
        }
        __syncthreads();                                       // A tile visible

        // ---- A fragment from LDS: row l16, 2x8 consecutive K -> 4x ds_load_b128
        const float* arow = &Abuf[c & 1][l16 * 32];
        float4 a0 = *(const float4*)(arow + half * 8);
        float4 a1 = *(const float4*)(arow + half * 8 + 4);
        float4 a2 = *(const float4*)(arow + 16 + half * 8);
        float4 a3 = *(const float4*)(arow + 16 + half * 8 + 4);
        union { v16bf v; __bf16 h[16]; } A;
        A.h[0] = (__bf16)a0.x;  A.h[1] = (__bf16)a0.y;
        A.h[2] = (__bf16)a0.z;  A.h[3] = (__bf16)a0.w;
        A.h[4] = (__bf16)a1.x;  A.h[5] = (__bf16)a1.y;
        A.h[6] = (__bf16)a1.z;  A.h[7] = (__bf16)a1.w;
        A.h[8] = (__bf16)a2.x;  A.h[9] = (__bf16)a2.y;
        A.h[10] = (__bf16)a2.z; A.h[11] = (__bf16)a2.w;
        A.h[12] = (__bf16)a3.x; A.h[13] = (__bf16)a3.y;
        A.h[14] = (__bf16)a3.z; A.h[15] = (__bf16)a3.w;

        // ---- B fragment: clamped unconditional loads + cndmask (branch-free)
        union { v16bf v; __bf16 h[16]; } Bv;
        #pragma unroll
        for (int j = 0; j < 16; ++j) {
            int k  = k0 + ((j >> 3) << 4) + (half << 3) + (j & 7);
            bool kv = (k < K);
            int kc = kv ? k : 0;
            int ci = kc / KK2;                    // compile-time divisor
            int kr = kc - ci * KK2;
            int ky = kr / KS;
            int kx = kr - ky * KS;
            int iy = oy * stride - pad + ky;
            int ix = ox * stride - pad + kx;
            bool v = kv && nV && (iy >= 0) && (iy < Hin) && (ix >= 0) && (ix < Win);
            int iyc = min(max(iy, 0), Hin - 1);
            int ixc = min(max(ix, 0), Win - 1);
            float f = inb[((long long)ci * Hin + iyc) * Win + ixc];
            Bv.h[j] = (__bf16)(v ? f : 0.f);
        }

        if (c + 1 < nchunks)   // hint next chunk's input row toward the caches
            __builtin_prefetch(inb + ((long long)((k0 + 32) / KK2) * Hin + oy) * Win + ox, 0, 1);

        acc = __builtin_amdgcn_wmma_f32_16x16x32_bf16(
                  false, A.v, false, Bv.v, (short)0, acc, false, false);
        __syncthreads();                                       // buf reusable
    }

    if (nV) {
        float* outb = out + (long long)b * out_bstride;
        #pragma unroll
        for (int r = 0; r < 8; ++r) {
            int m = m0 + r + half * 8;
            if (m < Cout) {
                float v = acc[r] + bias[m];
                if (leaky) v = (v >= 0.f) ? v : 0.1f * v;
                outb[(long long)m * npx + n] = v;
            }
        }
    }
}

// ---------------------------------------------------------------------------
// 81-channel cost volume (mean over channels, zero-padded shifts).
// ---------------------------------------------------------------------------
__global__ __launch_bounds__(256)
void corr_kernel(const float* __restrict__ x1, const float* __restrict__ x2,
                 float* __restrict__ out, int C, int H, int W,
                 long long x_bstride, long long out_bstride)
{
    int idx = blockIdx.x * blockDim.x + threadIdx.x;
    int npx = H * W;
    int b = blockIdx.z;
    if (idx >= npx) return;
    int y = idx / W, x = idx - y * W;
    const float* p1 = x1 + (long long)b * x_bstride;
    const float* p2 = x2 + (long long)b * x_bstride;
    float inv = 1.f / (float)C;
    float* ob = out + (long long)b * out_bstride;
    int o = 0;
    for (int dy = -4; dy <= 4; ++dy) {
        for (int dx = -4; dx <= 4; ++dx, ++o) {
            int yy = y + dy, xx = x + dx;
            float s = 0.f;
            if (yy >= 0 && yy < H && xx >= 0 && xx < W) {
                int off2 = yy * W + xx;
                for (int c = 0; c < C; ++c)
                    s += p1[(long long)c * npx + idx] * p2[(long long)c * npx + off2];
            }
            ob[(long long)o * npx + idx] = s * inv;
        }
    }
}

// ---------------------------------------------------------------------------
// Bilinear warp with zeros padding (per-corner validity, like grid_sample).
// ---------------------------------------------------------------------------
__global__ __launch_bounds__(256)
void warp_kernel(const float* __restrict__ x, const float* __restrict__ flow,
                 float* __restrict__ out, int C, int H, int W)
{
    int npx = H * W;
    int t = blockIdx.x * blockDim.x + threadIdx.x;
    if (t >= C * npx) return;
    int b = blockIdx.z;
    int c = t / npx;
    int idx = t - c * npx;
    int y = idx / W, xc = idx - y * W;

    const float* fb = flow + (long long)b * 2 * npx;
    float fx = (float)xc + fb[idx];
    float fy = (float)y  + fb[npx + idx];
    float x0f = floorf(fx), y0f = floorf(fy);
    float wx = fx - x0f, wy = fy - y0f;
    int x0 = (int)x0f, y0 = (int)y0f;

    const float* xb = x + ((long long)b * C + c) * npx;
    float v = 0.f;
    #pragma unroll
    for (int cy = 0; cy < 2; ++cy) {
        #pragma unroll
        for (int cx = 0; cx < 2; ++cx) {
            int xi = x0 + cx, yi = y0 + cy;
            float wgt = (cx ? wx : 1.f - wx) * (cy ? wy : 1.f - wy);
            bool ok = (xi >= 0) && (xi < W) && (yi >= 0) && (yi < H);
            int xic = min(max(xi, 0), W - 1);
            int yic = min(max(yi, 0), H - 1);
            float px = xb[yic * W + xic];
            v += ok ? wgt * px : 0.f;
        }
    }
    out[((long long)b * C + c) * npx + idx] = v;
}

// ---------------------------------------------------------------------------
// 2x bilinear upsample, align_corners=True, with pre-scale (flow *= 2).
// ---------------------------------------------------------------------------
__global__ __launch_bounds__(256)
void upsample2x_kernel(const float* __restrict__ in, float* __restrict__ out,
                       int C, int Hin, int Win, float scale)
{
    int Hout = 2 * Hin, Wout = 2 * Win;
    int npo = Hout * Wout, npi = Hin * Win;
    int t = blockIdx.x * blockDim.x + threadIdx.x;
    if (t >= C * npo) return;
    int b = blockIdx.z;
    int c = t / npo;
    int idx = t - c * npo;
    int y = idx / Wout, x = idx - y * Wout;

    float cy = (float)y * ((float)(Hin - 1) / (float)(Hout - 1));
    float cx = (float)x * ((float)(Win - 1) / (float)(Wout - 1));
    int y0 = (int)floorf(cy);  int y1 = min(y0 + 1, Hin - 1);
    int x0 = (int)floorf(cx);  int x1 = min(x0 + 1, Win - 1);
    float wy = cy - (float)y0, wx = cx - (float)x0;

    const float* ib = in + ((long long)b * C + c) * npi;
    float v00 = ib[y0 * Win + x0], v01 = ib[y0 * Win + x1];
    float v10 = ib[y1 * Win + x0], v11 = ib[y1 * Win + x1];
    float v = (v00 * (1.f - wx) + v01 * wx) * (1.f - wy)
            + (v10 * (1.f - wx) + v11 * wx) * wy;
    out[((long long)b * C + c) * npo + idx] = scale * v;
}

__global__ __launch_bounds__(256)
void zero_kernel(float* __restrict__ p, long long n)
{
    long long t = (long long)blockIdx.x * blockDim.x + threadIdx.x;
    if (t < n) p[t] = 0.f;
}

__global__ __launch_bounds__(256)
void add_kernel(float* __restrict__ a, const float* __restrict__ b, long long n)
{
    long long t = (long long)blockIdx.x * blockDim.x + threadIdx.x;
    if (t < n) a[t] += b[t];
}

__global__ __launch_bounds__(256)
void strided_copy_kernel(const float* __restrict__ in, long long in_bstride,
                         float* __restrict__ out, long long out_bstride, long long cnt)
{
    long long t = (long long)blockIdx.x * blockDim.x + threadIdx.x;
    int b = blockIdx.z;
    if (t < cnt) out[(long long)b * out_bstride + t] = in[(long long)b * in_bstride + t];
}

// ---------------------------------------------------------------------------
// Host orchestration
// ---------------------------------------------------------------------------
extern "C" void kernel_launch(void* const* d_in, const int* in_sizes, int n_in,
                              void* d_out, int out_size, void* d_ws, size_t ws_size,
                              hipStream_t stream)
{
    (void)in_sizes; (void)n_in; (void)ws_size;
    const int B = 4;
    // levels coarsest-first (matches pyramid loop order of reference)
    const int Hl[4] = {32, 64, 128, 256};
    const int Wl[4] = {56, 112, 224, 448};
    const int Cl[4] = {128, 96, 64, 32};
    const long long pxl[4] = {32LL*56, 64LL*112, 128LL*224, 256LL*448};
    const long long PXF = 256LL * 448;

    const float* seq1 = (const float*)d_in[0];
    const float* seq2 = (const float*)d_in[1];
    struct ConvP { const float* w; const float* b; };
    int ii = 2;
    ConvP enc[4][2];
    for (int l = 0; l < 4; ++l)
        for (int j = 0; j < 2; ++j) {
            enc[l][j].w = (const float*)d_in[ii++];
            enc[l][j].b = (const float*)d_in[ii++];
        }
    ConvP fe[4];
    for (int j = 0; j < 4; ++j) { fe[j].w = (const float*)d_in[ii++]; fe[j].b = (const float*)d_in[ii++]; }
    ConvP p1[4];
    for (int j = 0; j < 4; ++j) { p1[j].w = (const float*)d_in[ii++]; p1[j].b = (const float*)d_in[ii++]; }

    // ---- workspace layout (float offsets)
    float* ws = (float*)d_ws;
    size_t off = 0;
    auto alloc = [&](size_t n) { size_t o = off; off += n; return o; };
    size_t f1o[4], f2o[4];
    for (int l = 0; l < 4; ++l) f1o[l] = alloc((size_t)B * Cl[l] * pxl[l]);
    for (int l = 0; l < 4; ++l) f2o[l] = alloc((size_t)B * Cl[l] * pxl[l]);
    size_t tmpo  = alloc((size_t)B * 32 * PXF);
    size_t warpo = alloc((size_t)B * 32 * PXF);
    size_t cato  = alloc((size_t)B * 307 * PXF);
    size_t fao   = alloc((size_t)B * 2 * PXF);
    size_t fbo   = alloc((size_t)B * 2 * PXF);

    auto conv = [&](const float* in, long long inb, float* out, long long outb,
                    ConvP p, int Cin, int Cout, int Hin, int Win,
                    int Hout, int Wout, int s, int ks, int leaky) {
        dim3 g((unsigned)((Hout * (long long)Wout + 127) / 128),
               (unsigned)((Cout + 15) / 16), (unsigned)B);
        if (ks == 3)
            conv2d_wmma<3><<<g, 256, 0, stream>>>(in, inb, p.w, p.b, out, outb,
                                                  Cin, Cout, Hin, Win, Hout, Wout, s, leaky);
        else
            conv2d_wmma<1><<<g, 256, 0, stream>>>(in, inb, p.w, p.b, out, outb,
                                                  Cin, Cout, Hin, Win, Hout, Wout, s, leaky);
    };

    // ---- encoder (both images)
    for (int img = 0; img < 2; ++img) {
        const float* cur = img ? seq2 : seq1;
        long long curb = 25LL * 256 * 448;
        int Cin = 25, Hin = 256, Win = 448;
        for (int e = 0; e < 4; ++e) {
            int lv = 3 - e;
            int Cout = Cl[lv], Hout = Hl[lv], Wout = Wl[lv];
            int s = (e == 0) ? 1 : 2;
            long long hw = (long long)Hout * Wout;
            float* tmp = ws + tmpo;
            float* ft  = ws + (img ? f2o[lv] : f1o[lv]);
            conv(cur, curb, tmp, Cout * hw, enc[e][0], Cin, Cout, Hin, Win, Hout, Wout, s, 3, 1);
            conv(tmp, Cout * hw, ft, Cout * hw, enc[e][1], Cout, Cout, Hout, Wout, Hout, Wout, 1, 3, 1);
            cur = ft; curb = Cout * hw; Cin = Cout; Hin = Hout; Win = Wout;
        }
    }

    // ---- coarse-to-fine flow loop
    float* fa = ws + fao;
    float* fb = ws + fbo;
    float* flow = fa;
    for (int l = 0; l < 4; ++l) {
        int C = Cl[l], H = Hl[l], W = Wl[l];
        long long npx = (long long)H * W;
        const float* x1 = ws + f1o[l];
        const float* x2 = ws + f2o[l];
        const float* x2w;
        if (l == 0) {
            long long n = (long long)B * 2 * npx;
            zero_kernel<<<(unsigned)((n + 255) / 256), 256, 0, stream>>>(flow, n);
            x2w = x2;
        } else {
            float* nf = (flow == fa) ? fb : fa;
            long long no = (long long)2 * npx;
            upsample2x_kernel<<<dim3((unsigned)((no + 255) / 256), 1, B), 256, 0, stream>>>(
                flow, nf, 2, H / 2, W / 2, 2.0f);
            flow = nf;
            long long nw = (long long)C * npx;
            warp_kernel<<<dim3((unsigned)((nw + 255) / 256), 1, B), 256, 0, stream>>>(
                x2, flow, ws + warpo, C, H, W);
            x2w = ws + warpo;
        }
        float* cat = ws + cato;
        long long catb = 307LL * npx;
        corr_kernel<<<dim3((unsigned)((npx + 255) / 256), 1, B), 256, 0, stream>>>(
            x1, x2w, cat + 192 * npx, C, H, W, (long long)C * npx, catb);
        conv(x1, (long long)C * npx, cat + 273 * npx, catb, p1[l], C, 32, H, W, H, W, 1, 1, 0);
        strided_copy_kernel<<<dim3((unsigned)((2 * npx + 255) / 256), 1, B), 256, 0, stream>>>(
            flow, 2 * npx, cat + 305 * npx, catb, 2 * npx);
        conv(cat + 192 * npx, catb, cat + 96 * npx, catb, fe[0], 115, 96, H, W, H, W, 1, 3, 1);
        conv(cat +  96 * npx, catb, cat + 32 * npx, catb, fe[1], 211, 64, H, W, H, W, 1, 3, 1);
        conv(cat +  32 * npx, catb, cat +  0 * npx, catb, fe[2], 275, 32, H, W, H, W, 1, 3, 1);
        conv(cat, catb, ws + tmpo, 2 * npx, fe[3], 307, 2, H, W, H, W, 1, 3, 0);
        long long n = (long long)B * 2 * npx;
        add_kernel<<<(unsigned)((n + 255) / 256), 256, 0, stream>>>(flow, ws + tmpo, n);
    }

    hipMemcpyAsync(d_out, flow, (size_t)out_size * sizeof(float),
                   hipMemcpyDeviceToDevice, stream);
}
